// Mamba_model_6588479832613
// MI455X (gfx1250) — matrix operand order repaired and checked by reading, not compiled, not run on hardware
//
#include <hip/hip_runtime.h>
#include <hip/hip_bf16.h>
#include <math.h>

#define D_MODEL   512
#define D_INNER   1024
#define D_STATE   32
#define D_CONV    4
#define NHEADS    16
#define HEADDIM   64
#define CONV_DIM  1088      // D_INNER + 2*D_STATE
#define D_IN_PROJ 2128      // 2*D_INNER + 2*D_STATE + NHEADS
#define BATCH     2
#define SEQLEN    8000
#define MTOT      (BATCH*SEQLEN)   // 16000
#define EPSF      1e-5f

typedef unsigned short u16;
typedef __attribute__((ext_vector_type(16))) __bf16        v16bf;
typedef __attribute__((ext_vector_type(8)))  float         v8f;
typedef __attribute__((ext_vector_type(4)))  unsigned int  u32x4;

__device__ __forceinline__ u16 f2bf(float f) {
  unsigned u = __float_as_uint(f);
  u += 0x7fffu + ((u >> 16) & 1u);          // round-to-nearest-even
  return (u16)(u >> 16);
}
__device__ __forceinline__ float sigmoidf_(float x){ return 1.f/(1.f+__expf(-x)); }
__device__ __forceinline__ float softplusf_(float x){ return (x>20.f)?x:log1pf(__expf(x)); }

// gfx1250 async global->LDS DMA (tracked by ASYNCcnt), 16B per lane
__device__ __forceinline__ void async_ld_b128(unsigned lds_addr, const u16* gaddr){
  asm volatile("global_load_async_to_lds_b128 %0, %1, off"
               :: "v"(lds_addr), "v"(gaddr) : "memory");
}
__device__ __forceinline__ void wait_async0(){
  asm volatile("s_wait_asynccnt 0x0" ::: "memory");
}

// ---------------------------------------------------------------- conversions
__global__ void k_f32_to_bf16(const float* __restrict__ in, u16* __restrict__ out, int n){
  int i = blockIdx.x*blockDim.x + threadIdx.x;
  if(i < n) out[i] = f2bf(in[i]);
}

// ------------------------------------------------------------- front-end conv
__global__ void k_conv_front(const float* __restrict__ x,
                             const float* __restrict__ w1, const float* __restrict__ w2,
                             const float* __restrict__ w3, const float* __restrict__ w4,
                             float* __restrict__ y, float* __restrict__ stats){
  int c  = blockIdx.x;            // 0..511
  int br = c >> 7, ch = c & 127;
  const float* w; int k;
  if(br==0){ w=w1; k=1; } else if(br==1){ w=w2; k=5; }
  else if(br==2){ w=w3; k=9; } else { w=w4; k=13; }
  w += ch*k;
  int pad = (k-1)>>1;
  int idx = blockIdx.y*blockDim.x + threadIdx.x;   // 0..MTOT-1
  float v = 0.f;
  if(idx < MTOT){
    int b = idx / SEQLEN, l = idx % SEQLEN;
    const float* xb = x + (size_t)b*SEQLEN;
    for(int j=0;j<k;++j){
      int p = l + j - pad;
      float xv = (p>=0 && p<SEQLEN) ? xb[p] : 0.f;
      v += w[j]*xv;
    }
    y[(size_t)idx*D_MODEL + c] = v;
  }
  __shared__ float s1[256], s2[256];
  s1[threadIdx.x] = (idx<MTOT)? v    : 0.f;
  s2[threadIdx.x] = (idx<MTOT)? v*v  : 0.f;
  __syncthreads();
  for(int o=128;o>0;o>>=1){
    if(threadIdx.x<o){ s1[threadIdx.x]+=s1[threadIdx.x+o]; s2[threadIdx.x]+=s2[threadIdx.x+o]; }
    __syncthreads();
  }
  if(threadIdx.x==0){ atomicAdd(&stats[c], s1[0]); atomicAdd(&stats[512+c], s2[0]); }
}

// ------------------------------------------------- BN + ReLU + LayerNorm(512)
__global__ void k_bn_ln(float* __restrict__ y, u16* __restrict__ y16,
                        const float* __restrict__ stats,
                        const float* __restrict__ bng, const float* __restrict__ bnb,
                        const float* __restrict__ lnw, const float* __restrict__ lnb){
  int idx = blockIdx.x;          // position 0..MTOT-1
  int t   = threadIdx.x;         // 256
  const float invN = 1.f/(float)MTOT;
  float r[2];
  #pragma unroll
  for(int j=0;j<2;++j){
    int c = t + j*256;
    float v   = y[(size_t)idx*D_MODEL + c];
    float m   = stats[c]*invN;
    float var = fmaxf(stats[512+c]*invN - m*m, 0.f);
    float rr  = (v-m)*rsqrtf(var+EPSF)*bng[c] + bnb[c];
    r[j] = fmaxf(rr, 0.f);
  }
  __shared__ float s1[256], s2[256];
  s1[t] = r[0]+r[1]; s2[t] = r[0]*r[0]+r[1]*r[1];
  __syncthreads();
  for(int o=128;o>0;o>>=1){ if(t<o){ s1[t]+=s1[t+o]; s2[t]+=s2[t+o]; } __syncthreads(); }
  float mu  = s1[0]*(1.f/(float)D_MODEL);
  float var = fmaxf(s2[0]*(1.f/(float)D_MODEL) - mu*mu, 0.f);
  float inv = rsqrtf(var+EPSF);
  #pragma unroll
  for(int j=0;j<2;++j){
    int c = t + j*256;
    float o = (r[j]-mu)*inv*lnw[c] + lnb[c];
    y  [(size_t)idx*D_MODEL + c] = o;
    y16[(size_t)idx*D_MODEL + c] = f2bf(o);
  }
}

// ------------------------------------------------------------------ WMMA GEMM
// C[M,N] f32 = A[M,K] bf16 (row-major) * W[N,K]^T bf16 (row-major over K).
// WG tile 128x128, 8 waves in 4(M) x 2(N); each wave 32x64 = 2x4 16x16 WMMAs.
// LDS tiles double-buffered and filled by async global->LDS DMA.
#define TM 128
#define TN 128
__global__ void __launch_bounds__(256)
k_gemm_bf16(const u16* __restrict__ A, const u16* __restrict__ W,
            float* __restrict__ C, int M, int N, int K){
  __shared__ u16 As[2][TM][40];   // 32 K-cols + 8 pad (row = 80B, 16B-aligned)
  __shared__ u16 Bs[2][TN][40];
  int t    = threadIdx.x;
  int lane = t & 31, w = t >> 5;
  int wm   = w & 3,  wn = w >> 2;          // 4 x 2 waves
  int Mbase = blockIdx.y * TM;
  int Nbase = blockIdx.x * TN;
  v8f acc[2][4] = {};

  // tile staging assignment: thread -> (row r, 16-elt half), 256 thr = 128x32 tile
  int r    = t >> 1, half = t & 1;
  int arow = Mbase + r; if(arow >= M) arow = M-1;   // clamp addr; dead values
  int brow = Nbase + r; if(brow >= N) brow = N-1;   // are masked by store guard
  const u16* aptr = A + (size_t)arow*K + half*16;
  const u16* bptr = W + (size_t)brow*K + half*16;
  unsigned lA[2], lB[2];
  lA[0] = (unsigned)(unsigned long long)&As[0][r][half*16];
  lA[1] = (unsigned)(unsigned long long)&As[1][r][half*16];
  lB[0] = (unsigned)(unsigned long long)&Bs[0][r][half*16];
  lB[1] = (unsigned)(unsigned long long)&Bs[1][r][half*16];

  auto stage = [&](int buf, int kk){
    async_ld_b128(lA[buf],       aptr + kk);
    async_ld_b128(lA[buf] + 16u, aptr + kk + 8);
    async_ld_b128(lB[buf],       bptr + kk);
    async_ld_b128(lB[buf] + 16u, bptr + kk + 8);
  };

  stage(0, 0);
  wait_async0();
  __syncthreads();

  int buf = 0;
  for(int kk=0; kk<K; kk+=32){
    if(kk + 32 < K) stage(buf^1, kk + 32);   // DMA next tile during compute

    // A fragment: lanes 0-15: row=lane, K 0-7 & 16-23; lanes 16-31: K 8-15 & 24-31
    union Frag { v16bf v; u32x4 q[2]; } a[2], bfr[4];
    int sub = lane & 15;
    int akb = (lane < 16) ? 0 : 8;
    #pragma unroll
    for(int i=0;i<2;++i){
      int row = wm*32 + i*16 + sub;
      a[i].q[0] = *(const u32x4*)&As[buf][row][akb];
      a[i].q[1] = *(const u32x4*)&As[buf][row][akb+16];
    }
    // B fragment: lane n=lane%16; lanes 0-15 K 0-7 & 16-23; lanes 16-31: K 8-15 & 24-31
    int bkb = (lane < 16) ? 0 : 8;
    #pragma unroll
    for(int c=0;c<4;++c){
      int n = wn*64 + c*16 + sub;
      bfr[c].q[0] = *(const u32x4*)&Bs[buf][n][bkb];
      bfr[c].q[1] = *(const u32x4*)&Bs[buf][n][bkb+16];
    }
    #pragma unroll
    for(int i=0;i<2;++i)
      #pragma unroll
      for(int c=0;c<4;++c)
        acc[i][c] = __builtin_amdgcn_wmma_f32_16x16x32_bf16(
            false, a[i].v, false, bfr[c].v, (short)0, acc[i][c], false, false);

    wait_async0();      // my async writes into buf^1 have landed
    __syncthreads();    // everyone done reading buf & sees buf^1
    buf ^= 1;
  }

  // C/D layout: VGPR i -> M = i (+8 for lanes 16-31), N = lane%16
  int rbase = (lane < 16) ? 0 : 8;
  int col0  = lane & 15;
  #pragma unroll
  for(int i=0;i<2;++i){
    #pragma unroll
    for(int c=0;c<4;++c){
      int col = Nbase + wn*64 + c*16 + col0;
      if(col < N){
        #pragma unroll
        for(int e=0;e<8;++e){
          int row = Mbase + wm*32 + i*16 + rbase + e;
          if(row < M) C[(size_t)row*N + col] = acc[i][c][e];
        }
      }
    }
  }
}

// ----------------------------------------------- depthwise causal conv + SiLU
__global__ void k_dwconv_silu(const float* __restrict__ zx, const float* __restrict__ cw,
                              const float* __restrict__ cb, float* __restrict__ xbc){
  int gid = blockIdx.x*blockDim.x + threadIdx.x;
  if(gid >= MTOT*CONV_DIM) return;
  int c   = gid % CONV_DIM;
  int idx = gid / CONV_DIM;
  int b = idx / SEQLEN, l = idx % SEQLEN;
  float v = 0.f;
  #pragma unroll
  for(int j=0;j<D_CONV;++j){
    int p = l - (D_CONV-1) + j;
    if(p >= 0)
      v += cw[c*D_CONV + j] * zx[((size_t)(b*SEQLEN + p))*D_IN_PROJ + D_INNER + c];
  }
  v += cb[c];
  xbc[(size_t)idx*CONV_DIM + c] = v * sigmoidf_(v);
}

// --------------------------------------------------------------- state scan
__global__ void __launch_bounds__(256)
k_scan(const float* __restrict__ zx, const float* __restrict__ xbc,
       const float* __restrict__ dtb, const float* __restrict__ Alog,
       float* __restrict__ ys){
  int b = blockIdx.x >> 4;
  int h = blockIdx.x & 15;
  int t = threadIdx.x;
  int p = t >> 2, sub = t & 3, n0 = sub*8;
  float A      = -__expf(Alog[h]);
  float dtbias = dtb[h];
  __shared__ float xs[HEADDIM], Bsh[D_STATE], Csh[D_STATE];
  float hreg[8];
  #pragma unroll
  for(int j=0;j<8;++j) hreg[j]=0.f;

  for(int l=0;l<SEQLEN;++l){
    int idx = b*SEQLEN + l;
    const float* xrow = xbc + (size_t)idx*CONV_DIM;
    if(t < HEADDIM)                  xs [t]                 = xrow[h*HEADDIM + t];
    else if(t < HEADDIM + D_STATE)   Bsh[t-HEADDIM]         = xrow[D_INNER + (t-HEADDIM)];
    else if(t < HEADDIM + 2*D_STATE) Csh[t-HEADDIM-D_STATE] = xrow[D_INNER + D_STATE + (t-HEADDIM-D_STATE)];
    float dt = softplusf_(zx[(size_t)idx*D_IN_PROJ + (2*D_INNER + 2*D_STATE) + h] + dtbias);
    __syncthreads();
    float da   = __expf(dt*A);
    float coef = dt * xs[p];
    float acc  = 0.f;
    #pragma unroll
    for(int j=0;j<8;++j){
      hreg[j] = hreg[j]*da + coef*Bsh[n0+j];
      acc    += hreg[j]*Csh[n0+j];
    }
    acc += __shfl_xor(acc, 1);
    acc += __shfl_xor(acc, 2);
    if(sub == 0) ys[(size_t)idx*D_INNER + h*HEADDIM + p] = acc;
    __syncthreads();
  }
}

// ----------------------------------------- +D*x, SiLU(z) gate, RMSNorm(1024)
__global__ void k_gate_rmsnorm(const float* __restrict__ ys, const float* __restrict__ xbc,
                               const float* __restrict__ zx, const float* __restrict__ Dh,
                               const float* __restrict__ nw, u16* __restrict__ g16){
  int idx = blockIdx.x;
  int t   = threadIdx.x;
  float v[4]; float ss = 0.f;
  #pragma unroll
  for(int j=0;j<4;++j){
    int e = t + j*256;
    float xh = xbc[(size_t)idx*CONV_DIM + e];
    float y  = ys [(size_t)idx*D_INNER + e] + Dh[e>>6]*xh;
    float z  = zx [(size_t)idx*D_IN_PROJ + e];
    y *= z * sigmoidf_(z);
    v[j] = y; ss += y*y;
  }
  __shared__ float s1[256];
  s1[t] = ss; __syncthreads();
  for(int o=128;o>0;o>>=1){ if(t<o) s1[t]+=s1[t+o]; __syncthreads(); }
  float inv = rsqrtf(s1[0]*(1.f/(float)D_INNER) + EPSF);
  #pragma unroll
  for(int j=0;j<4;++j){
    int e = t + j*256;
    g16[(size_t)idx*D_INNER + e] = f2bf(v[j]*inv*nw[e]);
  }
}

// -------------------------------------------------------------- output head
__global__ void k_head(const float* __restrict__ y, const float* __restrict__ w,
                       const float* __restrict__ bias, float* __restrict__ out){
  int idx = blockIdx.x*blockDim.x + threadIdx.x;
  if(idx >= MTOT) return;
  const float* row = y + (size_t)idx*D_MODEL;
  float s = 0.f;
  for(int d=0; d<D_MODEL; ++d) s += row[d]*w[d];
  out[idx] = tanhf(s + bias[0]);
}

// ===========================================================================
extern "C" void kernel_launch(void* const* d_in, const int* in_sizes, int n_in,
                              void* d_out, int out_size, void* d_ws, size_t ws_size,
                              hipStream_t stream){
  const float* x      = (const float*)d_in[0];
  const float* cw1    = (const float*)d_in[1];
  const float* cw2    = (const float*)d_in[2];
  const float* cw3    = (const float*)d_in[3];
  const float* cw4    = (const float*)d_in[4];
  const float* bng    = (const float*)d_in[5];
  const float* bnb    = (const float*)d_in[6];
  const float* lnw    = (const float*)d_in[7];
  const float* lnb    = (const float*)d_in[8];
  const float* inproj = (const float*)d_in[9];    // (4, 2128, 512)
  const float* convw  = (const float*)d_in[10];   // (4, 1088, 1, 4)
  const float* convb  = (const float*)d_in[11];   // (4, 1088)
  const float* dtb    = (const float*)d_in[12];   // (4, 16)
  const float* Alog   = (const float*)d_in[13];   // (4, 16)
  const float* Dh     = (const float*)d_in[14];   // (4, 16)
  const float* normw  = (const float*)d_in[15];   // (4, 1024)
  const float* outw   = (const float*)d_in[16];   // (4, 512, 1024)
  const float* outpw  = (const float*)d_in[17];   // (1, 512)
  const float* outpb  = (const float*)d_in[18];   // (1,)
  float* out = (float*)d_out;

  char* ws = (char*)d_ws;
  size_t off = 0;
  auto carve = [&](size_t bytes)->void*{
    void* p = ws + off; off = (off + bytes + 255) & ~(size_t)255; return p;
  };
  float* buf_y    = (float*)carve((size_t)MTOT*D_MODEL*4);     //  32.8 MB
  u16*   buf_y16  = (u16*)  carve((size_t)MTOT*D_MODEL*2);     //  16.4 MB
  float* buf_zx   = (float*)carve((size_t)MTOT*D_IN_PROJ*4);   // 136.2 MB
  float* buf_xbc  = (float*)carve((size_t)MTOT*CONV_DIM*4);    //  69.6 MB
  float* buf_ys   = (float*)carve((size_t)MTOT*D_INNER*4);     //  65.5 MB
  u16*   buf_g16  = (u16*)  carve((size_t)MTOT*D_INNER*2);     //  32.8 MB
  u16*   buf_wi16 = (u16*)  carve((size_t)D_IN_PROJ*D_MODEL*2);//   2.2 MB
  u16*   buf_wo16 = (u16*)  carve((size_t)D_MODEL*D_INNER*2);  //   1.0 MB
  float* stats    = (float*)carve(1024*4);

  hipMemsetAsync(stats, 0, 1024*sizeof(float), stream);

  // front-end: 4 conv branches + BN + LN
  dim3 gconv(D_MODEL, (MTOT + 255)/256);
  k_conv_front<<<gconv, 256, 0, stream>>>(x, cw1, cw2, cw3, cw4, buf_y, stats);
  k_bn_ln<<<MTOT, 256, 0, stream>>>(buf_y, buf_y16, stats, bng, bnb, lnw, lnb);

  for(int i=0;i<4;++i){
    // in_proj: zxbcdt = y @ Wi^T   (M=16000, N=2128, K=512)
    int nwi = D_IN_PROJ*D_MODEL;
    k_f32_to_bf16<<<(nwi+255)/256, 256, 0, stream>>>(inproj + (size_t)i*nwi, buf_wi16, nwi);
    dim3 g1((D_IN_PROJ + TN - 1)/TN, MTOT/TM);
    k_gemm_bf16<<<g1, 256, 0, stream>>>(buf_y16, buf_wi16, buf_zx, MTOT, D_IN_PROJ, D_MODEL);

    // depthwise causal conv + SiLU on xBC slice
    int ndw = MTOT*CONV_DIM;
    k_dwconv_silu<<<(ndw+255)/256, 256, 0, stream>>>(
        buf_zx, convw + (size_t)i*CONV_DIM*D_CONV, convb + (size_t)i*CONV_DIM, buf_xbc);

    // selective scan
    k_scan<<<BATCH*NHEADS, 256, 0, stream>>>(
        buf_zx, buf_xbc, dtb + i*NHEADS, Alog + i*NHEADS, buf_ys);

    // gate + RMSNorm -> bf16
    k_gate_rmsnorm<<<MTOT, 256, 0, stream>>>(
        buf_ys, buf_xbc, buf_zx, Dh + i*NHEADS, normw + (size_t)i*D_INNER, buf_g16);

    // out_proj: y = g @ Wo^T  (M=16000, N=512, K=1024)
    int nwo = D_MODEL*D_INNER;
    k_f32_to_bf16<<<(nwo+255)/256, 256, 0, stream>>>(outw + (size_t)i*nwo, buf_wo16, nwo);
    dim3 g2((D_MODEL + TN - 1)/TN, MTOT/TM);
    k_gemm_bf16<<<g2, 256, 0, stream>>>(buf_g16, buf_wo16, buf_y, MTOT, D_MODEL, D_INNER);

    // bf16 copy of new y for next layer's in_proj
    int ny = MTOT*D_MODEL;
    k_f32_to_bf16<<<(ny+255)/256, 256, 0, stream>>>(buf_y, buf_y16, ny);
  }

  k_head<<<(MTOT+255)/256, 256, 0, stream>>>(buf_y, outpw, outpb, out);
}
